// ResNetLSTM_24739011625317
// MI455X (gfx1250) — matrix operand order repaired and checked by reading, not compiled, not run on hardware
//
#include <hip/hip_runtime.h>
#include <hip/hip_bf16.h>

typedef __attribute__((ext_vector_type(16))) _Float16 v16h;
typedef __attribute__((ext_vector_type(8)))  float    v8f;

#define WIN 1024
#define NBATCH 8192
#define TT 64            // conv time window per thread
#define CONV_HALO 39     // 18 left + 21 pipeline lag

// ---------------------------------------------------------------------------
// BN-fold kernel: fold gamma/var/mu/beta into conv weights+bias, write to ws.
// ---------------------------------------------------------------------------
struct ConvDesc {
    const float *w, *b, *g, *be, *mu, *va;
    int cout, cink, woff, boff;
};
struct FoldArgs { ConvDesc c[12]; };

__global__ void fold_kernel(FoldArgs fa, float* __restrict__ W) {
    const int tid = threadIdx.x;
    for (int ci = 0; ci < 12; ++ci) {
        ConvDesc d = fa.c[ci];
        const int n = d.cout * d.cink;
        for (int idx = tid; idx < n; idx += blockDim.x) {
            int o = idx / d.cink;
            float s = d.g[o] * rsqrtf(d.va[o] + 1e-5f);
            W[d.woff + idx] = d.w[idx] * s;
        }
        if (tid < d.cout) {
            float s = d.g[tid] * rsqrtf(d.va[tid] + 1e-5f);
            W[d.boff + tid] = d.b[tid] * s + d.be[tid] - d.mu[tid] * s;
        }
    }
}

// ---------------------------------------------------------------------------
// Conv phase: register-systolic pipeline through one residual stage.
// Stage lag: c1 (k=8, pad 3/4) -> 4, c2 (k=5) -> 2, c3 (k=3) -> 1  == 7.
// At iteration with newest input time ti: xin ring = x[ti-7..ti];
// c1 computed at u=ti-4, c2 at v=ti-6, c3+shortcut at w=ti-7.
// ---------------------------------------------------------------------------
template<int CIN, int COUT, int C1W, int C1B, int C2W, int C2B,
         int C3W, int C3B, int SCW, int SCB>
struct Stage {
    float xin[CIN][8];
    float r1[COUT][5];
    float r2[COUT][3];

    __device__ inline void init() {
#pragma unroll
        for (int c = 0; c < CIN; ++c)
#pragma unroll
            for (int i = 0; i < 8; ++i) xin[c][i] = 0.f;
#pragma unroll
        for (int c = 0; c < COUT; ++c) {
#pragma unroll
            for (int i = 0; i < 5; ++i) r1[c][i] = 0.f;
#pragma unroll
            for (int i = 0; i < 3; ++i) r2[c][i] = 0.f;
        }
    }

    __device__ inline void step(const float* __restrict__ W,
                                const float (&in)[CIN], float (&out)[COUT]) {
#pragma unroll
        for (int c = 0; c < CIN; ++c) {
#pragma unroll
            for (int i = 0; i < 7; ++i) xin[c][i] = xin[c][i + 1];
            xin[c][7] = in[c];
        }
        float t1[COUT];
#pragma unroll
        for (int o = 0; o < COUT; ++o) {
            float a = W[C1B + o];
#pragma unroll
            for (int i = 0; i < CIN; ++i)
#pragma unroll
                for (int k = 0; k < 8; ++k)
                    a = fmaf(W[C1W + (o * CIN + i) * 8 + k], xin[i][k], a);
            t1[o] = fmaxf(a, 0.f);
        }
#pragma unroll
        for (int o = 0; o < COUT; ++o) {
#pragma unroll
            for (int i = 0; i < 4; ++i) r1[o][i] = r1[o][i + 1];
            r1[o][4] = t1[o];
        }
        float t2[COUT];
#pragma unroll
        for (int o = 0; o < COUT; ++o) {
            float a = W[C2B + o];
#pragma unroll
            for (int i = 0; i < COUT; ++i)
#pragma unroll
                for (int k = 0; k < 5; ++k)
                    a = fmaf(W[C2W + (o * COUT + i) * 5 + k], r1[i][k], a);
            t2[o] = fmaxf(a, 0.f);
        }
#pragma unroll
        for (int o = 0; o < COUT; ++o) {
#pragma unroll
            for (int i = 0; i < 2; ++i) r2[o][i] = r2[o][i + 1];
            r2[o][2] = t2[o];
        }
#pragma unroll
        for (int o = 0; o < COUT; ++o) {
            float a = W[C3B + o];
#pragma unroll
            for (int i = 0; i < COUT; ++i)
#pragma unroll
                for (int k = 0; k < 3; ++k)
                    a = fmaf(W[C3W + (o * COUT + i) * 3 + k], r2[i][k], a);
            float sc = W[SCB + o];
#pragma unroll
            for (int i = 0; i < CIN; ++i)
                sc = fmaf(W[SCW + o * CIN + i], xin[i][0], sc);
            out[o] = fmaxf(a + sc, 0.f);
        }
    }
};

// y layout in ws: per (slice-local b, t): 8 halves (ch0..4, 0,0,0) = 16 B
__global__ void conv_kernel(const float* __restrict__ x,
                            const float* __restrict__ W,
                            unsigned short* __restrict__ yws,
                            int b0, int Bn) {
    const int tid = blockIdx.x * blockDim.x + threadIdx.x;
    if (tid >= Bn * (WIN / TT)) return;
    const int b  = tid >> 4;            // slice-local element (WIN/TT == 16)
    const int w0 = (tid & 15) * TT;
    const int gb = b0 + b;
    const float* __restrict__ xp = x + (size_t)gb * 3 * WIN;

    Stage<3, 2,   0,  48,  50,  70,  72,  84,  86,  92> s1;
    Stage<2, 2,  94, 126, 128, 148, 150, 162, 164, 168> s2;
    Stage<2, 5, 170, 250, 255, 380, 385, 460, 465, 475> s3;
    s1.init(); s2.init(); s3.init();

    for (int ti = w0 - 18; ti <= w0 + TT + 20; ++ti) {
        float in0[3];
        const bool inr = (ti >= 0) && (ti < WIN);
#pragma unroll
        for (int c = 0; c < 3; ++c) in0[c] = inr ? xp[c * WIN + ti] : 0.f;
        float o1[2]; s1.step(W, in0, o1);
        float o2[2]; s2.step(W, o1, o2);
        float o3[5]; s3.step(W, o2, o3);
        const int w = ti - 21;
        if (w >= w0 && w < w0 + TT) {
            union { uint4 u; unsigned short s[8]; } pk;
#pragma unroll
            for (int c = 0; c < 5; ++c)
                pk.s[c] = __builtin_bit_cast(unsigned short, (_Float16)o3[c]);
            pk.s[5] = pk.s[6] = pk.s[7] = 0;
            *(uint4*)(yws + ((size_t)b * WIN + w) * 8) = pk.u;
        }
    }
}

// ---------------------------------------------------------------------------
// LSTM + MLP. One wave = 16 batch elements. Per step one
// v_wmma_f32_16x16x32_f16 computes g[16 elems][16 gates] = [x_t|h] * [Wih;Whh]^T
// Gate order (torch): i(0-3) f(4-7) g(8-11) o(12-15) along N (= lane%16).
// D layout: reg r, lanes0-15 -> M=r, lanes16-31 -> M=r+8, N=lane%16.
// ---------------------------------------------------------------------------
__device__ inline float fast_sig(float x) {        // v_exp + v_rcp
    return __builtin_amdgcn_rcpf(1.0f + __expf(-x));
}

__global__ void lstm_kernel(const uint4* __restrict__ yws,
                            const float* __restrict__ w_ih,   // [16][5]
                            const float* __restrict__ w_hh,   // [16][4]
                            const float* __restrict__ mw1,    // [192][4]
                            const float* __restrict__ mb1,    // [192]
                            const float* __restrict__ mw2,    // [192]
                            const float* __restrict__ mb2,    // [1]
                            float* __restrict__ out, int b0) {
    __shared__ alignas(16) float hbuf[4][16][4];

    const int lane   = threadIdx.x & 31;
    const int wslot  = threadIdx.x >> 5;
    const int tile   = blockIdx.x * 4 + wslot;
    const int n      = lane & 15;        // N (gate) in D-layout; also row id
    const int half16 = lane >> 4;
    const int j      = lane & 3;
    const int base   = lane & 16;
    const int bLocal = tile * 16 + n;

    // --- B matrix (constant): lane<16 holds K=0..15 (half h -> K=h),
    //     lane>=16 holds K=16..31 (all zero here). B[K][N]: K<5 -> w_ih[N][K],
    //     5<=K<9 -> w_hh[N][K-5].
    v16h bw;
#pragma unroll
    for (int i = 0; i < 16; ++i) bw[i] = (_Float16)0.f;
    if (half16 == 0) {
#pragma unroll
        for (int k = 0; k < 5; ++k) bw[k] = (_Float16)w_ih[n * 5 + k];
#pragma unroll
        for (int k = 0; k < 4; ++k) bw[5 + k] = (_Float16)w_hh[n * 4 + k];
    }

    // --- A matrix (persistent zeros outside updated slots).
    // lane<16: halves0..8 = K0..8 = (y0..y4, h0,h1,h2,h3); halves9..15 = K9..15 = 0
    // lane>=16: half0 = K16 = h3 (B zero there); rest = K17..31 = 0
    v16h a;
#pragma unroll
    for (int i = 0; i < 16; ++i) a[i] = (_Float16)0.f;

    float cReg[8];
#pragma unroll
    for (int r = 0; r < 8; ++r) cReg[r] = 0.f;
    _Float16 hA0 = 0, hA1 = 0, hA2 = 0, hA3 = 0;

    // per-lane activation selectors: g-gate lanes use tanh = 2*sig(2x)-1
    const bool  isG  = (n >= 8) && (n < 12);
    const float preM = isG ? 2.f : 1.f;
    const float pM   = isG ? 2.f : 1.f;
    const float pA   = isG ? -1.f : 0.f;
    const int idxF = (base | (j + 4))  << 2;
    const int idxG = (base | (j + 8))  << 2;
    const int idxO = (base | (j + 12)) << 2;

    // x_t prefetch ring (one global_load_b128 per step, 4 ahead; L2-resident)
    const uint4* __restrict__ yp = yws + (size_t)bLocal * WIN;
    uint4 ybuf[4];
#pragma unroll
    for (int i = 0; i < 4; ++i) ybuf[i] = yp[i];

    float4 hv = make_float4(0.f, 0.f, 0.f, 0.f);

#pragma unroll 4
    for (int t = 0; t < WIN; ++t) {
        union { uint4 u; _Float16 h[8]; } yv;
        yv.u = ybuf[t & 3];
        if (half16 == 0) {
            a[0] = yv.h[0]; a[1] = yv.h[1]; a[2] = yv.h[2];
            a[3] = yv.h[3]; a[4] = yv.h[4];
            a[5] = hA0; a[6] = hA1; a[7] = hA2; a[8] = hA3;
        } else {
            a[0] = hA3;
        }
        if (t + 4 < WIN) ybuf[t & 3] = yp[t + 4];

        v8f cz = {0.f, 0.f, 0.f, 0.f, 0.f, 0.f, 0.f, 0.f};
        v8f d = __builtin_amdgcn_wmma_f32_16x16x32_f16(
            false, a, false, bw, (short)0, cz, false, false);

        float hr[8];
#pragma unroll
        for (int r = 0; r < 8; ++r) {
            float g   = d[r];
            float s   = fast_sig(preM * g);
            float act = fmaf(s, pM, pA);          // sig for i/f/o lanes, tanh for g
            int   ai  = __builtin_bit_cast(int, act);
            float aF  = __builtin_bit_cast(float, __builtin_amdgcn_ds_bpermute(idxF, ai));
            float aG  = __builtin_bit_cast(float, __builtin_amdgcn_ds_bpermute(idxG, ai));
            float aO  = __builtin_bit_cast(float, __builtin_amdgcn_ds_bpermute(idxO, ai));
            float cn  = fmaf(aF, cReg[r], act * aG);   // sig(f)*c + sig(i)*tanh(g)
            cReg[r]   = cn;
            float tc  = fmaf(fast_sig(2.f * cn), 2.f, -1.f);   // tanh(c)
            hr[r]     = aO * tc;                                // sig(o)*tanh(c)
        }

        // repack h (valid in j-lanes of each D-reg) for next A via LDS
        if (n < 4) {
            const int m0 = 8 * half16;
#pragma unroll
            for (int r = 0; r < 8; ++r) hbuf[wslot][m0 + r][j] = hr[r];
        }
        __builtin_amdgcn_wave_barrier();    // DS ops are in-order per wave
        hv = *(const float4*)&hbuf[wslot][n][0];
        hA0 = (_Float16)hv.x; hA1 = (_Float16)hv.y;
        hA2 = (_Float16)hv.z; hA3 = (_Float16)hv.w;
    }

    // fused MLP head: logits = w2 . relu(W1 h + b1) + b2
    if (half16 == 0) {
        float acc = mb2[0];
        for (int q = 0; q < 192; ++q) {
            float pre = mb1[q];
            pre = fmaf(hv.x, mw1[q * 4 + 0], pre);
            pre = fmaf(hv.y, mw1[q * 4 + 1], pre);
            pre = fmaf(hv.z, mw1[q * 4 + 2], pre);
            pre = fmaf(hv.w, mw1[q * 4 + 3], pre);
            acc = fmaf(mw2[q], fmaxf(pre, 0.f), acc);
        }
        out[b0 + bLocal] = acc;
    }
}

// ---------------------------------------------------------------------------
// Host side: input-leaf mapping (auto-detect flattening), ws slicing, launches
// ---------------------------------------------------------------------------
extern "C" void kernel_launch(void* const* d_in, const int* in_sizes, int n_in,
                              void* d_out, int out_size, void* d_ws, size_t ws_size,
                              hipStream_t stream) {
    const float* L[128];
    for (int i = 0; i < n_in && i < 128; ++i) L[i] = (const float*)d_in[i];

    struct CP { const float *w, *b, *g, *be, *mu, *va; };
    CP cp[12];
    const float *X = nullptr, *wih = nullptr, *whh = nullptr;
    const float *mw1 = nullptr, *mb1 = nullptr, *mw2 = nullptr, *mb2 = nullptr;

    if (in_sizes[0] > 1000000) {
        if (n_in > 1 && in_sizes[1] == 48) {
            // Scheme B: x first; params in insertion order, conv = w,b,gamma,beta,mu,var
            X = L[0];
            for (int i = 0; i < 12; ++i) {
                int s = 1 + 6 * i;
                cp[i] = { L[s], L[s + 1], L[s + 2], L[s + 3], L[s + 4], L[s + 5] };
            }
            wih = L[73]; whh = L[74];
            mw1 = L[75]; mb1 = L[76]; mw2 = L[77]; mb2 = L[78];
        } else {
            // Scheme C: x first; params sorted-flatten (b,beta,gamma,mu,var,w)
            X = L[0];
            whh = L[1]; wih = L[2]; mb1 = L[3]; mb2 = L[4]; mw1 = L[5]; mw2 = L[6];
            for (int i = 0; i < 12; ++i) {
                int s = 7 + 6 * i;
                cp[i] = { L[s + 5], L[s + 0], L[s + 2], L[s + 1], L[s + 3], L[s + 4] };
            }
        }
    } else if (in_sizes[0] == 48) {
        // Scheme D: params first (insertion order), x last
        for (int i = 0; i < 12; ++i) {
            int s = 6 * i;
            cp[i] = { L[s], L[s + 1], L[s + 2], L[s + 3], L[s + 4], L[s + 5] };
        }
        wih = L[72]; whh = L[73];
        mw1 = L[74]; mb1 = L[75]; mw2 = L[76]; mb2 = L[77];
        X = L[78];
    } else {
        // Scheme A: sorted-flatten of whole dict: lstm, mlp, stages..., x
        whh = L[0]; wih = L[1]; mb1 = L[2]; mb2 = L[3]; mw1 = L[4]; mw2 = L[5];
        for (int i = 0; i < 12; ++i) {
            int s = 6 + 6 * i;
            cp[i] = { L[s + 5], L[s + 0], L[s + 2], L[s + 1], L[s + 3], L[s + 4] };
        }
        X = L[78];
    }

    // conv descriptor table: cout, cin*k, ws weight offset, ws bias offset
    static const int CD[12][4] = {
        {2, 24,   0,  48}, {2, 10,  50,  70}, {2,  6,  72,  84}, {2,  3,  86,  92},
        {2, 16,  94, 126}, {2, 10, 128, 148}, {2,  6, 150, 162}, {2,  2, 164, 168},
        {5, 16, 170, 250}, {5, 25, 255, 380}, {5, 15, 385, 460}, {5,  2, 465, 475}
    };
    FoldArgs fa;
    for (int i = 0; i < 12; ++i) {
        fa.c[i].w = cp[i].w;  fa.c[i].b = cp[i].b;  fa.c[i].g = cp[i].g;
        fa.c[i].be = cp[i].be; fa.c[i].mu = cp[i].mu; fa.c[i].va = cp[i].va;
        fa.c[i].cout = CD[i][0]; fa.c[i].cink = CD[i][1];
        fa.c[i].woff = CD[i][2]; fa.c[i].boff = CD[i][3];
    }

    float* W = (float*)d_ws;
    unsigned short* Y = (unsigned short*)((char*)d_ws + 8192);
    const size_t perElem = (size_t)WIN * 16;   // 16 B per (b,t)
    const size_t avail = (ws_size > 8192) ? ws_size - 8192 : 0;
    int Bs = 64;
    static const int cands[8] = {8192, 4096, 2048, 1024, 512, 256, 128, 64};
    for (int i = 0; i < 8; ++i)
        if ((size_t)cands[i] * perElem <= avail) { Bs = cands[i]; break; }

    fold_kernel<<<1, 256, 0, stream>>>(fa, W);
    for (int b0 = 0; b0 < NBATCH; b0 += Bs) {
        conv_kernel<<<Bs / 16, 256, 0, stream>>>(X, W, Y, b0, Bs);
        lstm_kernel<<<Bs / 64, 128, 0, stream>>>((const uint4*)Y, wih, whh,
                                                 mw1, mb1, mw2, mb2,
                                                 (float*)d_out, b0);
    }
}
